// ProkBertAttention_86655260164779
// MI455X (gfx1250) — matrix operand order repaired and checked by reading, not compiled, not run on hardware
//
#include <hip/hip_runtime.h>
#include <hip/hip_bf16.h>
#include <math.h>

typedef __attribute__((ext_vector_type(16))) _Float16 v16h;
typedef __attribute__((ext_vector_type(8)))  _Float16 v8h;
typedef __attribute__((ext_vector_type(8)))  float    v8f;

constexpr int cB = 4;
constexpr int cS = 2048;
constexpr int cH = 12;
constexpr int cD = 64;
constexpr int QT = 64;     // queries per block (4 waves x 16)
constexpr int KC = 32;     // keys per chunk
constexpr int WIN = 64;    // WINDOW/2
constexpr float SCALE = 0.125f;   // 1/sqrt(64)
constexpr float NEGV = -1e9f;

__device__ __forceinline__ v16h cat8(v8h lo, v8h hi) {
    v16h r;
#pragma unroll
    for (int i = 0; i < 8; ++i) { r[i] = lo[i]; r[i + 8] = hi[i]; }
    return r;
}

__device__ __forceinline__ v8f wmma_f16(v16h a, v16h b, v8f c) {
    return __builtin_amdgcn_wmma_f32_16x16x32_f16(false, a, false, b, (short)0, c,
                                                  false, false);
}

__global__ __launch_bounds__(128)
void prokbert_attn_kernel(const float* __restrict__ qkv,
                          const float* __restrict__ cosp,
                          const float* __restrict__ sinp,
                          float* __restrict__ out) {
    const int qt   = blockIdx.x;
    const int h    = blockIdx.y;
    const int b    = blockIdx.z;
    const int q0   = qt * QT;
    const int tid  = threadIdx.x;
    const int lane = tid & 31;
    const int wave = tid >> 5;

    // LDS staging (f16). Row strides are multiples of 8 f16 = 16B for aligned b128 reads.
    __shared__ __align__(16) _Float16 qs[QT][72];      // Q, rope'd, row-major
    __shared__ __align__(16) _Float16 ks[KC][72];      // K, rope'd, row-major
    __shared__ __align__(16) _Float16 vts[cD][40];     // V transposed: [d][key]
    __shared__ __align__(16) _Float16 ps[4][16][40];   // per-wave P tile (16q x 32k)

    const long long sstride = 3LL * cH * cD;                         // qkv stride in s
    const long long hbase   = ((long long)(b * cS) * 3 * cH + h) * cD; // (b, s=0, t=0, h)

    // ---- Stage Q with RoPE, convert to f16 ----
    for (int i = tid; i < QT * 32; i += 128) {
        int row = i >> 5, dp = i & 31;
        int qi = q0 + row;
        long long base = hbase + (long long)qi * sstride;   // t=0 (Q)
        float x1 = qkv[base + dp];
        float x2 = qkv[base + dp + 32];
        float c1 = cosp[qi * cD + dp],      s1 = sinp[qi * cD + dp];
        float c2 = cosp[qi * cD + dp + 32], s2 = sinp[qi * cD + dp + 32];
        qs[row][dp]      = (_Float16)(x1 * c1 - x2 * s1);
        qs[row][dp + 32] = (_Float16)(x2 * c2 + x1 * s2);
    }
    __syncthreads();

    // ---- Q into A-operand registers (16x32 f16 layout), two d-slabs ----
    v16h qa0, qa1;
    {
        int m  = lane & 15;
        int kb = (lane & 16) ? 16 : 0;
        const _Float16* r = &qs[wave * 16 + m][0];
        qa0 = cat8(*(const v8h*)(r + kb),      *(const v8h*)(r + kb + 8));
        qa1 = cat8(*(const v8h*)(r + kb + 32), *(const v8h*)(r + kb + 40));
    }

    // ---- online-softmax state ----
    float mstate[8], lstate[8];
#pragma unroll
    for (int r = 0; r < 8; ++r) { mstate[r] = -__builtin_inff(); lstate[r] = 0.f; }
    v8f o0 = {}, o1 = {}, o2 = {}, o3 = {};

    const int n     = lane & 15;
    const int hisel = (lane & 16) ? 1 : 0;

    for (int c = 0; c < 6; ++c) {
        int kbase = q0 - WIN + c * KC;
        if (kbase + KC <= 0) continue;    // uniformly out of range (below 0)
        if (kbase >= cS)     continue;    // uniformly out of range (above S)

        // prefetch next chunk's K rows into caches
        if (tid < KC) {
            int kn = kbase + KC + tid;
            kn = kn < 0 ? 0 : (kn >= cS ? cS - 1 : kn);
            __builtin_prefetch(&qkv[hbase + (long long)kn * sstride + cH * cD], 0, 3);
        }

        // ---- K chunk: RoPE + f16 -> LDS row-major ----
        for (int i = tid; i < KC * 32; i += 128) {
            int row = i >> 5, dp = i & 31;
            int kk = kbase + row;
            int kc = kk < 0 ? 0 : (kk >= cS ? cS - 1 : kk);
            long long base = hbase + (long long)kc * sstride + cH * cD;   // t=1 (K)
            float x1 = qkv[base + dp];
            float x2 = qkv[base + dp + 32];
            float c1 = cosp[kc * cD + dp],      s1 = sinp[kc * cD + dp];
            float c2 = cosp[kc * cD + dp + 32], s2 = sinp[kc * cD + dp + 32];
            ks[row][dp]      = (_Float16)(x1 * c1 - x2 * s1);
            ks[row][dp + 32] = (_Float16)(x2 * c2 + x1 * s2);
        }
        // ---- V chunk: f16, transposed -> LDS [d][key] ----
        for (int i = tid; i < KC * cD; i += 128) {
            int row = i >> 6, d = i & 63;                 // coalesced along d
            int kk = kbase + row;
            int kc = kk < 0 ? 0 : (kk >= cS ? cS - 1 : kk);
            float v = qkv[hbase + (long long)kc * sstride + 2 * cH * cD + d]; // t=2 (V)
            vts[d][row] = (_Float16)v;
        }
        __syncthreads();

        // ---- B operands from LDS ----
        // scores B: column = key, contraction dim = d (contiguous per lane)
        int dsel = hisel ? 16 : 0;
        v16h kb00, kb01, kb10, kb11;
        {
            const _Float16* r0 = &ks[n][0];
            const _Float16* r1 = &ks[16 + n][0];
            kb00 = cat8(*(const v8h*)(r0 + dsel),      *(const v8h*)(r0 + dsel + 8));
            kb01 = cat8(*(const v8h*)(r0 + 32 + dsel), *(const v8h*)(r0 + 32 + dsel + 8));
            kb10 = cat8(*(const v8h*)(r1 + dsel),      *(const v8h*)(r1 + dsel + 8));
            kb11 = cat8(*(const v8h*)(r1 + 32 + dsel), *(const v8h*)(r1 + 32 + dsel + 8));
        }
        // P*V B: column = d, contraction dim = key (contiguous per lane in vts)
        v16h vb[4];
        {
            int ksel = hisel ? 16 : 0;
#pragma unroll
            for (int dt = 0; dt < 4; ++dt) {
                const _Float16* r = &vts[dt * 16 + n][0];
                vb[dt] = cat8(*(const v8h*)(r + ksel), *(const v8h*)(r + ksel + 8));
            }
        }

        // ---- S = Q K^T  (two 16-key halves, accumulate over d slabs) ----
        v8f s0 = {}, s1 = {};
        s0 = wmma_f16(qa0, kb00, s0);
        s0 = wmma_f16(qa1, kb01, s0);
        s1 = wmma_f16(qa0, kb10, s1);
        s1 = wmma_f16(qa1, kb11, s1);

        // ---- mask + online softmax (rows live in 16-lane half-waves) ----
        int k0 = kbase + n, k1 = kbase + 16 + n;
        int qrowbase = q0 + wave * 16 + (hisel ? 8 : 0);
#pragma unroll
        for (int r = 0; r < 8; ++r) {
            int qi = qrowbase + r;
            int d0 = qi - k0, d1 = qi - k1;
            bool ok0 = ((unsigned)k0 < (unsigned)cS) && (d0 <= WIN) && (d0 >= -WIN);
            bool ok1 = ((unsigned)k1 < (unsigned)cS) && (d1 <= WIN) && (d1 >= -WIN);
            float a  = s0[r] * SCALE + (ok0 ? 0.f : NEGV);
            float bb = s1[r] * SCALE + (ok1 ? 0.f : NEGV);
            float mx = fmaxf(a, bb);
            mx = fmaxf(mx, __shfl_xor(mx, 1, 32));
            mx = fmaxf(mx, __shfl_xor(mx, 2, 32));
            mx = fmaxf(mx, __shfl_xor(mx, 4, 32));
            mx = fmaxf(mx, __shfl_xor(mx, 8, 32));
            float mn = fmaxf(mstate[r], mx);
            float co = __expf(mstate[r] - mn);
            float p0 = __expf(a - mn), p1 = __expf(bb - mn);
            float sum = p0 + p1;
            sum += __shfl_xor(sum, 1, 32);
            sum += __shfl_xor(sum, 2, 32);
            sum += __shfl_xor(sum, 4, 32);
            sum += __shfl_xor(sum, 8, 32);
            lstate[r] = lstate[r] * co + sum;
            mstate[r] = mn;
            o0[r] *= co; o1[r] *= co; o2[r] *= co; o3[r] *= co;
            int prow = (hisel ? 8 : 0) + r;
            ps[wave][prow][n]      = (_Float16)p0;
            ps[wave][prow][16 + n] = (_Float16)p1;
        }
        __syncthreads();   // P visible to own wave; all waves done reading K/V LDS

        // ---- P into A-operand layout, O += P * V ----
        v16h pa;
        {
            int m   = lane & 15;
            int kb2 = hisel ? 16 : 0;
            const _Float16* r = &ps[wave][m][0];
            pa = cat8(*(const v8h*)(r + kb2), *(const v8h*)(r + kb2 + 8));
        }
        o0 = wmma_f16(pa, vb[0], o0);
        o1 = wmma_f16(pa, vb[1], o1);
        o2 = wmma_f16(pa, vb[2], o2);
        o3 = wmma_f16(pa, vb[3], o3);
    }

    // ---- finalize: divide by l, write (B, S, H*D) fp32, coalesced ----
    {
        int qrowbase = q0 + wave * 16 + (hisel ? 8 : 0);
#pragma unroll
        for (int r = 0; r < 8; ++r) {
            float inv = 1.f / lstate[r];
            int row = qrowbase + r;
            long long o = ((long long)(b * cS + row)) * (cH * cD) + h * cD + n;
            out[o +  0] = o0[r] * inv;
            out[o + 16] = o1[r] * inv;
            out[o + 32] = o2[r] * inv;
            out[o + 48] = o3[r] * inv;
        }
    }
}

extern "C" void kernel_launch(void* const* d_in, const int* in_sizes, int n_in,
                              void* d_out, int out_size, void* d_ws, size_t ws_size,
                              hipStream_t stream) {
    const float* qkv  = (const float*)d_in[0];
    const float* cosp = (const float*)d_in[1];
    const float* sinp = (const float*)d_in[2];
    // d_in[3] (dense S x S mask, 16 MB) intentionally unused: the band mask is
    // computed inline from indices, saving ~16 MB of HBM traffic.
    float* out = (float*)d_out;
    dim3 grid(cS / QT, cH, cB);   // (32, 12, 4) blocks
    dim3 block(128);              // 4 waves of 32
    prokbert_attn_kernel<<<grid, block, 0, stream>>>(qkv, cosp, sinp, out);
}
